// SAMGRUCell_76699525972203
// MI455X (gfx1250) — matrix operand order repaired and checked by reading, not compiled, not run on hardware
//
#include <hip/hip_runtime.h>

// ---------------- problem constants ----------------
#define BATCH 16384
#define HDIM  256
#define FEATD 256
#define GXD   70
#define GYD   70
#define SWIN  2
#define KWIN  5   // 2*SW+1
#define NCTX  25  // KWIN*KWIN

// ---------------- CDNA5 WMMA types ----------------
typedef __attribute__((ext_vector_type(16))) __bf16 bf16x16;
typedef __attribute__((ext_vector_type(8)))  float  f32x8;
typedef __attribute__((ext_vector_type(4)))  int    i32x4;

union Frag16 { bf16x16 v; uint4 q[2]; };

#define WMMA_BF16(A, Bm, C) \
  __builtin_amdgcn_wmma_f32_16x16x32_bf16(false, (A), false, (Bm), (short)0, (C), false, false)

// ---------------- gfx1250 async global->LDS copy helpers ----------------
#if __has_builtin(__builtin_amdgcn_global_load_async_to_lds_b128)
#define HAVE_ASYNC_LDS 1
#endif

__device__ __forceinline__ void async_cp16(const unsigned short* g, unsigned short* l) {
#ifdef HAVE_ASYNC_LDS
  // builtin signature: (int4 addrspace(1)* src, int4 addrspace(3)* dst,
  //                     imm int offset, imm int cpol)
  __builtin_amdgcn_global_load_async_to_lds_b128(
      (__attribute__((address_space(1))) i32x4*)(g),
      (__attribute__((address_space(3))) i32x4*)(l), 0, 0);
#else
  *(uint4*)l = *(const uint4*)g;          // register relay fallback
#endif
}

template <int N>
__device__ __forceinline__ void wait_async() {
#ifdef HAVE_ASYNC_LDS
#if __has_builtin(__builtin_amdgcn_s_wait_asynccnt)
  __builtin_amdgcn_s_wait_asynccnt((short)N);
#else
  asm volatile("s_wait_asynccnt %0" ::"i"(N) : "memory");
#endif
#endif
}

// ---------------- math helpers ----------------
__device__ __forceinline__ unsigned short f2bf(float f) {   // f32->bf16 RNE
  unsigned int u = __float_as_uint(f);
  u += 0x7FFFu + ((u >> 16) & 1u);
  return (unsigned short)(u >> 16);
}
__device__ __forceinline__ float sigmoidf_(float x) {
  return 1.0f / (1.0f + __expf(-x));
}
__device__ __forceinline__ float tanhf_(float x) {          // overflow-safe fast tanh
  float a = fabsf(x);
  float e = __expf(-2.0f * a);                              // in (0,1]
  float t = (1.0f - e) / (1.0f + e);
  return copysignf(t, x);
}

// ---------------- fragment loaders ----------------
// A-fragment: 16-bit A matrix (16M x 32K). lane -> M = lane&15,
// elements 0..7 -> K = kbase + (lane>>4)*8 + j ; elements 8..15 -> +16.
__device__ __forceinline__ bf16x16 loadA(const unsigned short* __restrict__ rowp,
                                         int kbase, int hl) {
  Frag16 f;
  f.q[0] = *(const uint4*)(rowp + kbase + hl * 8);
  f.q[1] = *(const uint4*)(rowp + kbase + 16 + hl * 8);
  return f.v;
}

// B-fragment from an LDS-staged tile (16 N rows x ld K bf16, row-major):
// lane -> N = lane&15; elements 0..7 -> K = kbase + (lane>>4)*8 + j ; elements 8..15 -> +16.
__device__ __forceinline__ bf16x16 loadB_lds(const unsigned short* Lbase, int ld,
                                             int kbase, int lane) {
  const int n  = lane & 15;
  const int hl = lane >> 4;
  const unsigned short* p = Lbase + n * ld + kbase + hl * 8;
  Frag16 f;
  f.q[0] = *(const uint4*)(p);
  f.q[1] = *(const uint4*)(p + 16);
  return f.v;
}

// ---------------- stage 0: f32 -> bf16 converts ----------------
__global__ void k_cvt(const float* __restrict__ src, unsigned short* __restrict__ dst, int n) {
  int i = blockIdx.x * blockDim.x + threadIdx.x;
  if (i < n) dst[i] = f2bf(src[i]);
}

__global__ void k_cvt_feat(const float* __restrict__ input, unsigned short* __restrict__ dst) {
  int i = blockIdx.x * blockDim.x + threadIdx.x;  // 0 .. B*256
  int row = i >> 8, col = i & 255;
  dst[i] = f2bf(input[(size_t)row * (FEATD + 2) + col]);
}

// ---------------- stage 1: gate GEMMs + pointwise gates ----------------
// 128 threads = 4 waves; each wave owns 16 batch rows. All 4 waves share
// LDS-staged weight tiles: per c-tile, 8 tiles (wih/whh x 4 gate blocks) of
// 16x256 bf16 = 64 KB, double-buffered (128 KB dynamic LDS).
__global__ __launch_bounds__(128) void k_gates(
    const unsigned short* __restrict__ featB,   // B x 256 bf16
    const unsigned short* __restrict__ hidB,    // B x 256 bf16
    const unsigned short* __restrict__ wihB,    // 1024 x 256 bf16
    const unsigned short* __restrict__ whhB,    // 1024 x 256 bf16
    const float* __restrict__ b_ih,             // 1024
    const float* __restrict__ b_hh,             // 1024
    float* __restrict__ qf,                     // B x 256 (newgate)
    float* __restrict__ updf,                   // B x 256
    float* __restrict__ spatf,                  // B x 256
    unsigned short* __restrict__ comb)          // B x 512 bf16 ([mix | q])
{
  extern __shared__ unsigned short smem[];      // 2 x 32768 ushorts (2 x 64 KB)

  const int tid   = threadIdx.x;
  const int wave  = tid >> 5;
  const int lane  = tid & 31;
  const int mbase = (blockIdx.x * 4 + wave) * 16;
  const int mrow  = lane & 15;
  const int hl    = lane >> 4;

  // issue one c-tile's 8 weight tiles (64 KB) as 32 async b128s per thread
  auto issue_tiles = [&](int c, unsigned short* dstbase) {
#pragma unroll
    for (int i = 0; i < 32; ++i) {
      const int idx  = tid + i * 128;       // 0..4095 16B-chunks
      const int t    = idx >> 9;            // tile 0..7
      const int rem  = idx & 511;
      const int n    = rem >> 5;            // row-in-tile 0..15
      const int k16  = rem & 31;            // 16B chunk in row
      const int gate = t >> 1;              // 0..3 (r,i,n,s blocks)
      const unsigned short* W = (t & 1) ? whhB : wihB;
      const unsigned short* src = W + (size_t)(gate * HDIM + c + n) * FEATD + k16 * 8;
      async_cp16(src, dstbase + t * 4096 + n * 256 + k16 * 8);
    }
  };

  // resident A fragments for this wave's 16 rows (K = 256 -> 8 frags each)
  const unsigned short* fr = featB + (size_t)(mbase + mrow) * FEATD;
  const unsigned short* hr = hidB  + (size_t)(mbase + mrow) * HDIM;
  bf16x16 Af[8], Ah[8];
#pragma unroll
  for (int kk = 0; kk < 8; ++kk) {
    Af[kk] = loadA(fr, kk * 32, hl);
    Ah[kk] = loadA(hr, kk * 32, hl);
  }

  issue_tiles(0, smem);                      // prologue: fill buffer 0

  for (int ci = 0; ci < 16; ++ci) {
    const int c   = ci * 16;
    const int cur = ci & 1;
    unsigned short* buf = smem + cur * 32768;

    if (ci + 1 < 16) {                       // prefetch next tile set
      issue_tiles(c + 16, smem + (1 - cur) * 32768);
      wait_async<32>();                      // current buffer's copies done
    } else {
      wait_async<0>();
    }
    __syncthreads();                         // all waves' copies visible

    // tiles: 0=wih_r 1=whh_r 2=wih_i 3=whh_i 4=wih_n 5=whh_n 6=wih_s 7=whh_s
    f32x8 accR = {};
#pragma unroll
    for (int kk = 0; kk < 8; ++kk)
      accR = WMMA_BF16(Af[kk], loadB_lds(buf + 0 * 4096, 256, kk * 32, lane), accR);
#pragma unroll
    for (int kk = 0; kk < 8; ++kk)
      accR = WMMA_BF16(Ah[kk], loadB_lds(buf + 1 * 4096, 256, kk * 32, lane), accR);

    f32x8 accIN = {}, accHN = {};
#pragma unroll
    for (int kk = 0; kk < 8; ++kk)
      accIN = WMMA_BF16(Af[kk], loadB_lds(buf + 4 * 4096, 256, kk * 32, lane), accIN);
#pragma unroll
    for (int kk = 0; kk < 8; ++kk)
      accHN = WMMA_BF16(Ah[kk], loadB_lds(buf + 5 * 4096, 256, kk * 32, lane), accHN);

    f32x8 accI = {};
#pragma unroll
    for (int kk = 0; kk < 8; ++kk)
      accI = WMMA_BF16(Af[kk], loadB_lds(buf + 2 * 4096, 256, kk * 32, lane), accI);
#pragma unroll
    for (int kk = 0; kk < 8; ++kk)
      accI = WMMA_BF16(Ah[kk], loadB_lds(buf + 3 * 4096, 256, kk * 32, lane), accI);

    f32x8 accS = {};
#pragma unroll
    for (int kk = 0; kk < 8; ++kk)
      accS = WMMA_BF16(Af[kk], loadB_lds(buf + 6 * 4096, 256, kk * 32, lane), accS);
#pragma unroll
    for (int kk = 0; kk < 8; ++kk)
      accS = WMMA_BF16(Ah[kk], loadB_lds(buf + 7 * 4096, 256, kk * 32, lane), accS);

    const int col = c + (lane & 15);
    const float bR  = b_ih[col] + b_hh[col];
    const float bIN = b_ih[2 * HDIM + col];
    const float bHN = b_hh[2 * HDIM + col];
    const float bI  = b_ih[HDIM + col] + b_hh[HDIM + col];
    const float bS  = b_ih[3 * HDIM + col] + b_hh[3 * HDIM + col];

#pragma unroll
    for (int v = 0; v < 8; ++v) {
      const int row = mbase + hl * 8 + v;
      const float r  = sigmoidf_(accR[v] + bR);
      const float ng = tanhf_((accIN[v] + bIN) + r * (accHN[v] + bHN));
      const float ug = sigmoidf_(accI[v] + bI);
      const float sg = sigmoidf_(accS[v] + bS);
      const size_t o = (size_t)row * HDIM + col;
      qf[o]    = ng;
      updf[o]  = ug;
      spatf[o] = sg;
      comb[(size_t)row * (2 * HDIM) + HDIM + col] = f2bf(ng);
    }
    __syncthreads();                         // reads done before buffer reuse
  }
}

// ---------------- stage 2: gather + attention + mix ----------------
// one wave per batch row; each lane owns 8 contiguous channels
__global__ __launch_bounds__(256) void k_attn(
    const float* __restrict__ input,    // B x 258 (coords at cols 256,257)
    const float* __restrict__ qf,       // B x 256
    const float* __restrict__ memory,   // 70 x 70 x 256
    unsigned short* __restrict__ comb)  // B x 512; write cols [0,256) = bf16(mix)
{
  const int row  = blockIdx.x * 8 + (threadIdx.x >> 5);
  const int lane = threadIdx.x & 31;

  int gx = (int)input[(size_t)row * (FEATD + 2) + FEATD + 0] + SWIN;
  int gy = (int)input[(size_t)row * (FEATD + 2) + FEATD + 1] + SWIN;
  gx = min(max(gx, 0), GXD - 1);
  gy = min(max(gy, 0), GYD - 1);

  const float* qrp = qf + (size_t)row * HDIM + lane * 8;
  const float4 q0 = *(const float4*)(qrp);
  const float4 q1 = *(const float4*)(qrp + 4);

  float attn[NCTX];
#pragma unroll
  for (int k = 0; k < NCTX; ++k) {
    const int xi = min(max(gx + (k / KWIN) - SWIN, 0), GXD - 1);
    const int yi = min(max(gy + (k % KWIN) - SWIN, 0), GYD - 1);
    const float* ctx = memory + ((size_t)xi * GYD + yi) * HDIM + lane * 8;
    const float4 c0 = *(const float4*)(ctx);
    const float4 c1 = *(const float4*)(ctx + 4);
    float p = q0.x * c0.x + q0.y * c0.y + q0.z * c0.z + q0.w * c0.w +
              q1.x * c1.x + q1.y * c1.y + q1.z * c1.z + q1.w * c1.w;
#pragma unroll
    for (int s = 16; s > 0; s >>= 1) p += __shfl_xor(p, s, 32);
    attn[k] = (p == 0.0f) ? -__builtin_inff() : p;
  }

  float m = -__builtin_inff();
#pragma unroll
  for (int k = 0; k < NCTX; ++k) m = fmaxf(m, attn[k]);

  float4 m0 = {0.f, 0.f, 0.f, 0.f}, m1 = {0.f, 0.f, 0.f, 0.f};
  if (m != -__builtin_inff()) {   // else: softmax is all-NaN -> weights 0 -> mix 0
    float wsum = 0.0f;
    float w[NCTX];
#pragma unroll
    for (int k = 0; k < NCTX; ++k) { w[k] = __expf(attn[k] - m); wsum += w[k]; }
    const float inv = 1.0f / wsum;
#pragma unroll
    for (int k = 0; k < NCTX; ++k) {
      const int xi = min(max(gx + (k / KWIN) - SWIN, 0), GXD - 1);
      const int yi = min(max(gy + (k % KWIN) - SWIN, 0), GYD - 1);
      const float* ctx = memory + ((size_t)xi * GYD + yi) * HDIM + lane * 8;
      const float4 c0 = *(const float4*)(ctx);
      const float4 c1 = *(const float4*)(ctx + 4);
      const float wk = w[k] * inv;
      m0.x += wk * c0.x; m0.y += wk * c0.y; m0.z += wk * c0.z; m0.w += wk * c0.w;
      m1.x += wk * c1.x; m1.y += wk * c1.y; m1.z += wk * c1.z; m1.w += wk * c1.w;
    }
  }

  unsigned short* dst = comb + (size_t)row * (2 * HDIM) + lane * 8;
  dst[0] = f2bf(m0.x); dst[1] = f2bf(m0.y); dst[2] = f2bf(m0.z); dst[3] = f2bf(m0.w);
  dst[4] = f2bf(m1.x); dst[5] = f2bf(m1.y); dst[6] = f2bf(m1.z); dst[7] = f2bf(m1.w);
}

// ---------------- stage 3: output GEMM + final blend ----------------
// 128 threads = 4 waves x 16 rows. LDS-stages 16x512 bf16 w_out tiles
// (16 KB), double-buffered (32 KB dynamic LDS).
__global__ __launch_bounds__(128) void k_out(
    const unsigned short* __restrict__ comb,    // B x 512 bf16
    const unsigned short* __restrict__ woutB,   // 256 x 512 bf16
    const float* __restrict__ b_out,            // 256
    const float* __restrict__ qf,
    const float* __restrict__ spatf,
    const float* __restrict__ updf,
    const float* __restrict__ hidden,           // B x 256 f32 (original input)
    float* __restrict__ out)                    // B x 256 f32
{
  extern __shared__ unsigned short smem[];      // 2 x 8192 ushorts (2 x 16 KB)

  const int tid   = threadIdx.x;
  const int wave  = tid >> 5;
  const int lane  = tid & 31;
  const int mbase = (blockIdx.x * 4 + wave) * 16;
  const int mrow  = lane & 15;
  const int hl    = lane >> 4;

  auto issue_tile = [&](int nb, unsigned short* dstbase) {
#pragma unroll
    for (int i = 0; i < 8; ++i) {
      const int idx = tid + i * 128;       // 0..1023 16B-chunks
      const int n   = idx >> 6;            // row 0..15
      const int k16 = idx & 63;            // chunk in 1KB row
      const unsigned short* src = woutB + (size_t)(nb + n) * (2 * HDIM) + k16 * 8;
      async_cp16(src, dstbase + n * 512 + k16 * 8);
    }
  };

  const unsigned short* cr = comb + (size_t)(mbase + mrow) * (2 * HDIM);
  bf16x16 Ac[16];
#pragma unroll
  for (int kk = 0; kk < 16; ++kk) Ac[kk] = loadA(cr, kk * 32, hl);

  issue_tile(0, smem);

  for (int ni = 0; ni < 16; ++ni) {
    const int n   = ni * 16;
    const int cur = ni & 1;
    unsigned short* buf = smem + cur * 8192;

    if (ni + 1 < 16) {
      issue_tile(n + 16, smem + (1 - cur) * 8192);
      wait_async<8>();
    } else {
      wait_async<0>();
    }
    __syncthreads();

    f32x8 acc = {};
#pragma unroll
    for (int kk = 0; kk < 16; ++kk)
      acc = WMMA_BF16(Ac[kk], loadB_lds(buf, 512, kk * 32, lane), acc);

    const int col = n + (lane & 15);
    const float bias = b_out[col];
#pragma unroll
    for (int v = 0; v < 8; ++v) {
      const int row = mbase + hl * 8 + v;
      const size_t o = (size_t)row * HDIM + col;
      const float a   = tanhf_(acc[v] + bias);
      const float ng  = qf[o];
      const float cur_s = ng + spatf[o] * a;
      out[o] = cur_s + updf[o] * (hidden[o] - cur_s);
    }
    __syncthreads();
  }
}

// ---------------- launcher ----------------
extern "C" void kernel_launch(void* const* d_in, const int* in_sizes, int n_in,
                              void* d_out, int out_size, void* d_ws, size_t ws_size,
                              hipStream_t stream) {
  const float* input  = (const float*)d_in[0];  // B x 258
  const float* hidden = (const float*)d_in[1];  // B x 256
  const float* w_ih   = (const float*)d_in[2];  // 1024 x 256
  const float* b_ih   = (const float*)d_in[3];
  const float* w_hh   = (const float*)d_in[4];  // 1024 x 256
  const float* b_hh   = (const float*)d_in[5];
  const float* w_out  = (const float*)d_in[6];  // 256 x 512
  const float* b_out  = (const float*)d_in[7];
  const float* memory = (const float*)d_in[8];  // 70 x 70 x 256
  float* out = (float*)d_out;

  // workspace carve-up (all chunks 16B-aligned by construction)
  unsigned short* featB = (unsigned short*)d_ws;                  // B*256
  unsigned short* hidB  = featB + (size_t)BATCH * FEATD;          // B*256
  unsigned short* wihB  = hidB  + (size_t)BATCH * HDIM;           // 1024*256
  unsigned short* whhB  = wihB  + (size_t)4 * HDIM * FEATD;       // 1024*256
  unsigned short* woutB = whhB  + (size_t)4 * HDIM * HDIM;        // 256*512
  unsigned short* comb  = woutB + (size_t)HDIM * 2 * HDIM;        // B*512
  float* qf    = (float*)(comb + (size_t)BATCH * 2 * HDIM);       // B*256
  float* updf  = qf   + (size_t)BATCH * HDIM;                     // B*256
  float* spatf = updf + (size_t)BATCH * HDIM;                     // B*256

  // stage 0: bf16 conversions
  {
    int n;
    n = BATCH * FEATD;
    k_cvt_feat<<<(n + 255) / 256, 256, 0, stream>>>(input, featB);
    n = BATCH * HDIM;
    k_cvt<<<(n + 255) / 256, 256, 0, stream>>>(hidden, hidB, n);
    n = 4 * HDIM * FEATD;
    k_cvt<<<(n + 255) / 256, 256, 0, stream>>>(w_ih, wihB, n);
    n = 4 * HDIM * HDIM;
    k_cvt<<<(n + 255) / 256, 256, 0, stream>>>(w_hh, whhB, n);
    n = HDIM * 2 * HDIM;
    k_cvt<<<(n + 255) / 256, 256, 0, stream>>>(w_out, woutB, n);
  }

  // stage 1: gate GEMMs (each block: 4 waves x 16 rows = 64 rows, 128 KB LDS)
  k_gates<<<BATCH / 64, 128, 2 * 65536, stream>>>(featB, hidB, wihB, whhB,
                                                  b_ih, b_hh, qf, updf, spatf, comb);

  // stage 2: attention (8 rows per 256-thread block)
  k_attn<<<BATCH / 8, 256, 0, stream>>>(input, qf, memory, comb);

  // stage 3: output GEMM + blend (32 KB LDS)
  k_out<<<BATCH / 64, 128, 2 * 16384, stream>>>(comb, woutB, b_out, qf, spatf,
                                                updf, hidden, out);
}
